// GPTv3_83872121356494
// MI455X (gfx1250) — matrix operand order repaired and checked by reading, not compiled, not run on hardware
//
#include <hip/hip_runtime.h>
#include <hip/hip_bf16.h>
#include <math.h>

// ---------------------------------------------------------------------------
// Types / helpers
// ---------------------------------------------------------------------------
typedef __attribute__((ext_vector_type(16))) __bf16 v16bf;
typedef __attribute__((ext_vector_type(8)))  float  v8f;

union AB16 { v16bf v; unsigned u[8]; unsigned short s[16]; };

#define Dm    1024
#define Hn    16
#define HSd   64
#define Lseq  1024
#define Bb    4
#define NROWS 4096          // B*L
#define NLAY  12

__device__ __forceinline__ unsigned short f2bf(float f) {
    unsigned u = __float_as_uint(f);
    u += 0x7FFFu + ((u >> 16) & 1u);          // round-to-nearest-even
    return (unsigned short)(u >> 16);
}
__device__ __forceinline__ float bf2f(unsigned short s) {
    return __uint_as_float(((unsigned)s) << 16);
}
__device__ __forceinline__ unsigned pk2(float a, float b) {
    return (unsigned)f2bf(a) | ((unsigned)f2bf(b) << 16);
}
// WMMA 16x16x32 bf16 per-lane K patterns (ISA 7.12.2, wave32)
__device__ __forceinline__ int a_k(int v, int lane) {   // A operand (16x32)
    return 2 * v + ((v >= 4) ? 8 : 0) + ((lane & 16) ? 8 : 0);
}
__device__ __forceinline__ int b_k(int v, int lane) {   // B operand (32x16)
    return 2 * v + ((v >= 4) ? 8 : 0) + ((lane & 16) ? 8 : 0);
}
__device__ __forceinline__ float geluf(float x) {       // exact gelu
    return 0.5f * x * (1.0f + erff(x * 0.70710678118654752f));
}
// CDNA5 async global->LDS copy (ASYNCcnt-tracked, no VGPR round trip).
// LDS dest address = low 32 bits of the generic pointer.
__device__ __forceinline__ void async_copy_b128(void* lds_dst, const void* gsrc) {
    unsigned ldsaddr = (unsigned)(unsigned long long)(uintptr_t)lds_dst;
    unsigned long long gaddr = (unsigned long long)(uintptr_t)gsrc;
    asm volatile("global_load_async_to_lds_b128 %0, %1, off"
                 :: "v"(ldsaddr), "v"(gaddr) : "memory");
}
__device__ __forceinline__ void async_wait0() {
    asm volatile("s_wait_asynccnt 0" ::: "memory");
}

// ---------------------------------------------------------------------------
// Embedding gather + input projection:  x[4096,1024] = emb[row,156] @ proj_w + b
// ---------------------------------------------------------------------------
__global__ void embed_proj_kernel(const int* __restrict__ stage,
                                  const int* __restrict__ egoc,
                                  const int* __restrict__ oppc,
                                  const int* __restrict__ egoa,
                                  const int* __restrict__ oppa,
                                  const float* __restrict__ gamestate,
                                  const float* __restrict__ stage_emb,
                                  const float* __restrict__ char_emb,
                                  const float* __restrict__ action_emb,
                                  const float* __restrict__ proj_w,
                                  const float* __restrict__ proj_b,
                                  float* __restrict__ x)
{
    __shared__ float e[160];
    int row = blockIdx.x;
    int t = threadIdx.x;
    if (t < 156) {
        int c = t; float val;
        if (c < 4)               val = stage_emb[stage[row] * 4 + c];
        else if ((c -= 4) < 12)  val = char_emb[egoc[row] * 12 + c];
        else if ((c -= 12) < 12) val = char_emb[oppc[row] * 12 + c];
        else if ((c -= 12) < 32) val = action_emb[egoa[row] * 32 + c];
        else if ((c -= 32) < 32) val = action_emb[oppa[row] * 32 + c];
        else { c -= 32;          val = gamestate[(size_t)row * 64 + c]; }
        e[t] = val;
    }
    __syncthreads();
    for (int j = 0; j < 4; ++j) {
        int col = t + j * 256;
        float acc = proj_b[col];
        #pragma unroll 4
        for (int kk = 0; kk < 156; ++kk) acc += e[kk] * proj_w[kk * 1024 + col];
        x[(size_t)row * 1024 + col] = acc;
    }
}

// ---------------------------------------------------------------------------
// LayerNorm: one wave32 per row, butterfly reductions. bf16 or f32 output.
// ---------------------------------------------------------------------------
__global__ void ln_kernel(const float* __restrict__ x,
                          const float* __restrict__ w,
                          const float* __restrict__ b,
                          unsigned short* __restrict__ outB,
                          float* __restrict__ outF)
{
    int wid = threadIdx.x >> 5, lane = threadIdx.x & 31;
    int row = blockIdx.x * 4 + wid;
    const float* xr = x + (size_t)row * 1024;
    float r[32];
    float s = 0.f;
    #pragma unroll
    for (int j = 0; j < 32; ++j) { r[j] = xr[lane + j * 32]; s += r[j]; }
    #pragma unroll
    for (int m = 1; m < 32; m <<= 1) s += __shfl_xor(s, m, 32);
    float mean = s * (1.f / 1024.f);
    float vs = 0.f;
    #pragma unroll
    for (int j = 0; j < 32; ++j) { float d = r[j] - mean; vs += d * d; }
    #pragma unroll
    for (int m = 1; m < 32; m <<= 1) vs += __shfl_xor(vs, m, 32);
    float inv = rsqrtf(vs * (1.f / 1024.f) + 1e-5f);
    #pragma unroll
    for (int j = 0; j < 32; ++j) {
        int col = lane + j * 32;
        float o = (r[j] - mean) * inv * w[col] + b[col];
        if (outB) outB[(size_t)row * 1024 + col] = f2bf(o);
        else      outF[(size_t)row * 1024 + col] = o;
    }
}

// ---------------------------------------------------------------------------
// WMMA GEMM: C[M,N] = A_bf16[M,K] @ W_f32[K,N] + bias, fused epilogues.
//   mode 0: outF = resid + C            (f32, residual add)
//   mode 1: outB = bf16(C)              (plain bf16)
//   mode 2: outB = bf16(gelu(C))        (exact gelu, bf16)
// Block tile 128x128x32; 8 waves of 64x32.
//  - A tile staged with CDNA5 GLOBAL_LOAD_ASYNC_TO_LDS_B128 (no VGPR round trip)
//  - W staged f32->bf16, transposed [n][k], (k,k+1) pairs packed -> ds_store_b32
// ---------------------------------------------------------------------------
__global__ __launch_bounds__(256) void gemm_wmma_kernel(
    const unsigned short* __restrict__ A, const float* __restrict__ W,
    const float* __restrict__ bias, const float* __restrict__ resid,
    float* __restrict__ outF, unsigned short* __restrict__ outB,
    int M, int N, int K, int mode)
{
    __shared__ __align__(16) unsigned short lA[128][40];   // [m][k] bf16
    __shared__ __align__(16) unsigned short lB[128][40];   // [n][k] bf16 (transposed)
    int t = threadIdx.x;
    int lane = t & 31, wv = t >> 5;
    int wm = (wv >> 2) * 64, wn = (wv & 3) * 32;
    int m0 = blockIdx.y * 128, n0 = blockIdx.x * 128;

    v8f acc[4][2];
    #pragma unroll
    for (int i = 0; i < 4; ++i)
        #pragma unroll
        for (int j = 0; j < 2; ++j)
            #pragma unroll
            for (int v = 0; v < 8; ++v) acc[i][j][v] = 0.f;

    for (int k0 = 0; k0 < K; k0 += 32) {
        // stage A tile (bf16, 8KB): 512 x 16B chunks, 2 async copies per thread
        #pragma unroll
        for (int it = 0; it < 2; ++it) {
            int c = t + it * 256;
            int row = c >> 2, seg = c & 3;
            async_copy_b128(&lA[row][seg * 8],
                            A + (size_t)(m0 + row) * K + k0 + seg * 8);
        }
        // stage W tile (f32 -> bf16, transposed): two k-rows per task so the
        // (k,k+1) bf16 pair packs into one ds_store_b32
        #pragma unroll
        for (int it = 0; it < 2; ++it) {
            int c = t + it * 256;
            int kp = c >> 5, cs = c & 31;     // kp: 0..15 (k pair), cs: n/4 chunk
            int kk = kp * 2;
            const float* w0 = W + (size_t)(k0 + kk) * N + n0 + cs * 4;
            float4 a4 = *(const float4*)(w0);
            float4 b4 = *(const float4*)(w0 + N);
            *(unsigned*)&lB[cs * 4 + 0][kk] = pk2(a4.x, b4.x);
            *(unsigned*)&lB[cs * 4 + 1][kk] = pk2(a4.y, b4.y);
            *(unsigned*)&lB[cs * 4 + 2][kk] = pk2(a4.z, b4.z);
            *(unsigned*)&lB[cs * 4 + 3][kk] = pk2(a4.w, b4.w);
        }
        if (k0 + 32 < K) {   // prefetch next W tile -> global_prefetch_b8
            __builtin_prefetch(W + (size_t)(k0 + 32 + (t >> 3)) * N + n0 + (t & 7) * 16, 0, 1);
        }
        async_wait0();       // ASYNCcnt == 0 before the barrier
        __syncthreads();

        AB16 af[4], bfr[2];
        #pragma unroll
        for (int mt = 0; mt < 4; ++mt) {
            int mm = wm + mt * 16 + (lane & 15);
            #pragma unroll
            for (int v = 0; v < 8; ++v)
                af[mt].u[v] = *(const unsigned*)&lA[mm][a_k(v, lane)];
        }
        #pragma unroll
        for (int nt = 0; nt < 2; ++nt) {
            int nn = wn + nt * 16 + (lane & 15);
            #pragma unroll
            for (int v = 0; v < 8; ++v)
                bfr[nt].u[v] = *(const unsigned*)&lB[nn][b_k(v, lane)];
        }
        #pragma unroll
        for (int mt = 0; mt < 4; ++mt)
            #pragma unroll
            for (int nt = 0; nt < 2; ++nt)
                acc[mt][nt] = __builtin_amdgcn_wmma_f32_16x16x32_bf16(
                    false, af[mt].v, false, bfr[nt].v, (short)0, acc[mt][nt], false, false);
        __syncthreads();
    }

    // epilogue (C/D layout: row = v + 8*hi, col = lane&15)
    int hi = (lane & 16) ? 8 : 0;
    #pragma unroll
    for (int mt = 0; mt < 4; ++mt)
        #pragma unroll
        for (int nt = 0; nt < 2; ++nt)
            #pragma unroll
            for (int v = 0; v < 8; ++v) {
                int m = m0 + wm + mt * 16 + v + hi;
                int n = n0 + wn + nt * 16 + (lane & 15);
                float val = acc[mt][nt][v] + bias[n];
                size_t idx = (size_t)m * N + n;
                if (mode == 0)      outF[idx] = resid[idx] + val;
                else if (mode == 2) outB[idx] = f2bf(geluf(val));
                else                outB[idx] = f2bf(val);
            }
}

// ---------------------------------------------------------------------------
// Flash attention with Music-Transformer relative logits.
// Grid: B*H*(L/64) workgroups, 4 waves; each wave = one 16-row strip.
//   score[i,j] = (q_i.k_j + q_i.Er[1023-(i-j)]) / 8, causal, online softmax.
// P strip (rel logits) precomputed with WMMA into per-wave LDS (bf16).
// ---------------------------------------------------------------------------
__global__ __launch_bounds__(128) void attn_kernel(
    const unsigned short* __restrict__ qkv,   // [B*L, 3072] bf16 (q|k|v)
    const float* __restrict__ Er,             // [1024, 64] f32 (this layer)
    unsigned short* __restrict__ y)           // [B*L, 1024] bf16
{
    extern __shared__ unsigned short smem[];
    int lane = threadIdx.x & 31, wid = threadIdx.x >> 5;
    int iblk = blockIdx.x & 15;        // L/64 = 16
    int bh   = blockIdx.x >> 4;
    int b = bh >> 4, h = bh & 15;      // H = 16
    int i0 = iblk * 64 + wid * 16;
    int l15 = lane & 15, hi = (lane >> 4) & 1;

    unsigned short* P     = smem + (size_t)wid * (16 * 1024 + 16 * 32);
    unsigned short* probs = P + 16 * 1024;

    // q A-fragments (K = 64 -> 2 chunks of 32)
    AB16 qf[2];
    {
        size_t qbase = ((size_t)b * Lseq + i0 + l15) * 3072 + h * 64;
        #pragma unroll
        for (int c = 0; c < 2; ++c)
            #pragma unroll
            for (int v = 0; v < 8; ++v)
                qf[c].u[v] = *(const unsigned*)(qkv + qbase + a_k(v, lane) + c * 32);
    }

    // P[m][d] = q_{i0+m} . Er[1023-d], d in [0, i0+16)
    for (int t0 = 0; t0 < i0 + 16; t0 += 16) {
        int tt = t0 + l15;
        const float* er = Er + (size_t)(1023 - tt) * 64;
        AB16 eb[2];
        #pragma unroll
        for (int c = 0; c < 2; ++c)
            #pragma unroll
            for (int v = 0; v < 8; ++v) {
                int k = b_k(v, lane) + c * 32;
                float2 f = *(const float2*)(er + k);
                eb[c].u[v] = pk2(f.x, f.y);
            }
        v8f pa;
        #pragma unroll
        for (int v = 0; v < 8; ++v) pa[v] = 0.f;
        pa = __builtin_amdgcn_wmma_f32_16x16x32_bf16(false, qf[0].v, false, eb[0].v, (short)0, pa, false, false);
        pa = __builtin_amdgcn_wmma_f32_16x16x32_bf16(false, qf[1].v, false, eb[1].v, (short)0, pa, false, false);
        #pragma unroll
        for (int v = 0; v < 8; ++v) {
            int m = v + hi * 8;
            P[m * 1024 + t0 + l15] = f2bf(pa[v]);
        }
    }

    float M_run[8], L_run[8];
    v8f O[4];
    #pragma unroll
    for (int v = 0; v < 8; ++v) { M_run[v] = -INFINITY; L_run[v] = 0.f; }
    #pragma unroll
    for (int c = 0; c < 4; ++c)
        #pragma unroll
        for (int v = 0; v < 8; ++v) O[c][v] = 0.f;

    for (int j0 = 0; j0 <= i0 + 15; j0 += 32) {
        float sA[2][8];
        #pragma unroll
        for (int hf = 0; hf < 2; ++hf) {
            int jt = j0 + hf * 16;
            if (jt <= i0 + 15) {
                AB16 kb[2];
                size_t kbase = ((size_t)b * Lseq + jt + l15) * 3072 + 1024 + h * 64;
                #pragma unroll
                for (int c = 0; c < 2; ++c)
                    #pragma unroll
                    for (int v = 0; v < 8; ++v)
                        kb[c].u[v] = *(const unsigned*)(qkv + kbase + b_k(v, lane) + c * 32);
                v8f sc;
                #pragma unroll
                for (int v = 0; v < 8; ++v) sc[v] = 0.f;
                sc = __builtin_amdgcn_wmma_f32_16x16x32_bf16(false, qf[0].v, false, kb[0].v, (short)0, sc, false, false);
                sc = __builtin_amdgcn_wmma_f32_16x16x32_bf16(false, qf[1].v, false, kb[1].v, (short)0, sc, false, false);
                #pragma unroll
                for (int v = 0; v < 8; ++v) {
                    int m = v + hi * 8;
                    int i = i0 + m, j = jt + l15, d = i - j;
                    sA[hf][v] = (d >= 0) ? (sc[v] + bf2f(P[m * 1024 + d])) * 0.125f
                                         : -INFINITY;
                }
            } else {
                #pragma unroll
                for (int v = 0; v < 8; ++v) sA[hf][v] = -INFINITY;
            }
        }
        // online softmax over the 32-col chunk (rows live in 16-lane halves)
        #pragma unroll
        for (int v = 0; v < 8; ++v) {
            float tm = fmaxf(sA[0][v], sA[1][v]);
            #pragma unroll
            for (int msk = 1; msk < 16; msk <<= 1) tm = fmaxf(tm, __shfl_xor(tm, msk, 32));
            float mnew  = fmaxf(M_run[v], tm);
            float alpha = __expf(M_run[v] - mnew);
            float p0 = __expf(sA[0][v] - mnew);
            float p1 = __expf(sA[1][v] - mnew);
            float rs = p0 + p1;
            #pragma unroll
            for (int msk = 1; msk < 16; msk <<= 1) rs += __shfl_xor(rs, msk, 32);
            L_run[v] = L_run[v] * alpha + rs;
            M_run[v] = mnew;
            #pragma unroll
            for (int c = 0; c < 4; ++c) O[c][v] *= alpha;
            int m = v + hi * 8;
            probs[m * 32 + l15]      = f2bf(p0);
            probs[m * 32 + 16 + l15] = f2bf(p1);
        }
        // probs -> A fragment, then O += probs @ V  (4 WMMAs over HS=64)
        AB16 pf;
        #pragma unroll
        for (int v = 0; v < 8; ++v)
            pf.u[v] = *(const unsigned*)&probs[l15 * 32 + a_k(v, lane)];
        #pragma unroll
        for (int c = 0; c < 4; ++c) {
            AB16 vf;
            int ncol = 2048 + h * 64 + c * 16 + l15;
            #pragma unroll
            for (int v = 0; v < 8; ++v) {
                int k = b_k(v, lane);
                unsigned short u0 = qkv[((size_t)b * Lseq + j0 + k) * 3072 + ncol];
                unsigned short u1 = qkv[((size_t)b * Lseq + j0 + k + 1) * 3072 + ncol];
                vf.u[v] = (unsigned)u0 | ((unsigned)u1 << 16);
            }
            O[c] = __builtin_amdgcn_wmma_f32_16x16x32_bf16(false, pf.v, false, vf.v, (short)0, O[c], false, false);
        }
    }

    #pragma unroll
    for (int c = 0; c < 4; ++c)
        #pragma unroll
        for (int v = 0; v < 8; ++v) {
            int m = v + hi * 8;
            float val = O[c][v] / L_run[v];
            y[((size_t)b * Lseq + i0 + m) * 1024 + h * 64 + c * 16 + l15] = f2bf(val);
        }
}

// ---------------------------------------------------------------------------
// Output heads: btn[4096,17] | main[4096,21] | cstick[4096,21] (concat, f32)
// ---------------------------------------------------------------------------
__global__ void heads_kernel(const float* __restrict__ xf,
                             const float* __restrict__ btn_w,
                             const float* __restrict__ main_w,
                             const float* __restrict__ cstick_w,
                             float* __restrict__ out)
{
    __shared__ float xr[1024];
    int row = blockIdx.x, t = threadIdx.x;
    #pragma unroll
    for (int j = 0; j < 16; ++j) xr[t + j * 64] = xf[(size_t)row * 1024 + t + j * 64];
    __syncthreads();
    if (t < 59) {
        const float* W; int C, c; size_t base;
        if (t < 17)      { W = btn_w;    C = 17; c = t;      base = 0; }
        else if (t < 38) { W = main_w;   C = 21; c = t - 17; base = (size_t)4096 * 17; }
        else             { W = cstick_w; C = 21; c = t - 38; base = (size_t)4096 * 17 + (size_t)4096 * 21; }
        float acc = 0.f;
        for (int kk = 0; kk < 1024; ++kk) acc += xr[kk] * W[kk * C + c];
        out[base + (size_t)row * C + c] = acc;
    }
}

// ---------------------------------------------------------------------------
// Host-side orchestration
// ---------------------------------------------------------------------------
extern "C" void kernel_launch(void* const* d_in, const int* in_sizes, int n_in,
                              void* d_out, int out_size, void* d_ws, size_t ws_size,
                              hipStream_t stream)
{
    (void)in_sizes; (void)n_in; (void)out_size; (void)ws_size;
    const int*   stage      = (const int*)d_in[0];
    const int*   egoc       = (const int*)d_in[1];
    const int*   oppc       = (const int*)d_in[2];
    const int*   egoa       = (const int*)d_in[3];
    const int*   oppa       = (const int*)d_in[4];
    const float* gamestate  = (const float*)d_in[5];
    const float* stage_emb  = (const float*)d_in[6];
    const float* char_emb   = (const float*)d_in[7];
    const float* action_emb = (const float*)d_in[8];
    const float* proj_w     = (const float*)d_in[9];
    const float* proj_b     = (const float*)d_in[10];
    const float* ln1_w      = (const float*)d_in[11];
    const float* ln1_b      = (const float*)d_in[12];
    const float* attn_w     = (const float*)d_in[13];
    const float* attn_b     = (const float*)d_in[14];
    const float* attnproj_w = (const float*)d_in[15];
    const float* attnproj_b = (const float*)d_in[16];
    const float* Er         = (const float*)d_in[17];
    const float* ln2_w      = (const float*)d_in[18];
    const float* ln2_b      = (const float*)d_in[19];
    const float* fc_w       = (const float*)d_in[20];
    const float* fc_b       = (const float*)d_in[21];
    const float* mlpproj_w  = (const float*)d_in[22];
    const float* mlpproj_b  = (const float*)d_in[23];
    const float* lnf_w      = (const float*)d_in[24];
    const float* lnf_b      = (const float*)d_in[25];
    const float* btn_w      = (const float*)d_in[26];
    const float* main_w     = (const float*)d_in[27];
    const float* cstick_w   = (const float*)d_in[28];

    char* ws = (char*)d_ws;
    float*          x    = (float*)(ws);                                    // 16 MB residual
    unsigned short* xn   = (unsigned short*)(ws + (((size_t)16) << 20));    // 8 MB bf16
    unsigned short* qkvb = (unsigned short*)(ws + (((size_t)24) << 20));    // 24 MB bf16
    unsigned short* yb   = (unsigned short*)(ws + (((size_t)48) << 20));    // 8 MB bf16
    unsigned short* hb   = (unsigned short*)(ws + (((size_t)56) << 20));    // 32 MB bf16
    float*          xf   = (float*)(ws + (((size_t)88) << 20));             // 16 MB

    const size_t ATTN_LDS = 4 * (size_t)(16 * 1024 + 16 * 32) * sizeof(unsigned short);

    embed_proj_kernel<<<NROWS, 256, 0, stream>>>(stage, egoc, oppc, egoa, oppa,
        gamestate, stage_emb, char_emb, action_emb, proj_w, proj_b, x);

    for (int i = 0; i < NLAY; ++i) {
        ln_kernel<<<1024, 128, 0, stream>>>(x, ln1_w + (size_t)i * Dm,
                                            ln1_b + (size_t)i * Dm, xn, nullptr);
        gemm_wmma_kernel<<<dim3(3072 / 128, NROWS / 128), 256, 0, stream>>>(
            xn, attn_w + (size_t)i * Dm * 3072, attn_b + (size_t)i * 3072,
            nullptr, nullptr, qkvb, NROWS, 3072, Dm, 1);
        attn_kernel<<<Bb * Hn * (Lseq / 64), 128, ATTN_LDS, stream>>>(
            qkvb, Er + (size_t)i * 1024 * 64, yb);
        gemm_wmma_kernel<<<dim3(Dm / 128, NROWS / 128), 256, 0, stream>>>(
            yb, attnproj_w + (size_t)i * Dm * Dm, attnproj_b + (size_t)i * Dm,
            x, x, nullptr, NROWS, Dm, Dm, 0);
        ln_kernel<<<1024, 128, 0, stream>>>(x, ln2_w + (size_t)i * Dm,
                                            ln2_b + (size_t)i * Dm, xn, nullptr);
        gemm_wmma_kernel<<<dim3(4096 / 128, NROWS / 128), 256, 0, stream>>>(
            xn, fc_w + (size_t)i * Dm * 4096, fc_b + (size_t)i * 4096,
            nullptr, nullptr, hb, NROWS, 4096, Dm, 2);
        gemm_wmma_kernel<<<dim3(Dm / 128, NROWS / 128), 256, 0, stream>>>(
            hb, mlpproj_w + (size_t)i * 4096 * Dm, mlpproj_b + (size_t)i * Dm,
            x, x, nullptr, NROWS, Dm, 4096, 0);
    }

    ln_kernel<<<1024, 128, 0, stream>>>(x, lnf_w, lnf_b, nullptr, xf);
    heads_kernel<<<NROWS, 64, 0, stream>>>(xf, btn_w, main_w, cstick_w, (float*)d_out);
}
